// GraphEncoder_12257836662784
// MI455X (gfx1250) — compile-verified, matrix operand order checked by reading, not run on hardware
//
#include <hip/hip_runtime.h>
#include <hip/hip_bf16.h>
#include <stdint.h>

#define USE_ASYNC_LDS 1   // stage B tiles with global_load_async_to_lds_b128 (ASYNCcnt)

typedef __attribute__((ext_vector_type(16))) __bf16 v16bf;
typedef __attribute__((ext_vector_type(8)))  float  v8f;
typedef __attribute__((ext_vector_type(4)))  unsigned int v4u;

union Frag16 { v16bf v; v4u u[2]; };

static __device__ __forceinline__ v4u ld128(const void* p) {
    return *(const v4u*)p;
}

#if USE_ASYNC_LDS
static __device__ __forceinline__ void async_b128(unsigned ldsoff, const void* g) {
    unsigned long long ga = (unsigned long long)(size_t)g;
    asm volatile("global_load_async_to_lds_b128 %0, %1, off"
                 :: "v"(ldsoff), "v"(ga) : "memory");
}
#endif

// ---------------- small utility kernels ----------------

__global__ void k_fill_f32(float* p, float v, int n) {
    int i = blockIdx.x * blockDim.x + threadIdx.x;
    if (i < n) p[i] = v;
}

__global__ void k_deg_edges(const long long* __restrict__ ei, float* __restrict__ deg, int E) {
    int e = blockIdx.x * blockDim.x + threadIdx.x;
    if (e < E) atomicAdd(&deg[(int)ei[E + e]], 1.0f);
}

__global__ void k_rsqrt(const float* __restrict__ deg, float* __restrict__ dis, int n) {
    int i = blockIdx.x * blockDim.x + threadIdx.x;
    if (i < n) dis[i] = rsqrtf(deg[i]);
}

__global__ void k_t_self(const float* __restrict__ x, const float* __restrict__ dis,
                         float* __restrict__ t, int n) {
    int i = blockIdx.x * blockDim.x + threadIdx.x;
    if (i < n) { float d = dis[i]; t[i] = x[i] * d * d; }
}

__global__ void k_t_edges(const float* __restrict__ x, const long long* __restrict__ ei,
                          const float* __restrict__ dis, float* __restrict__ t, int E) {
    int e = blockIdx.x * blockDim.x + threadIdx.x;
    if (e < E) {
        int s = (int)ei[e], d = (int)ei[E + e];
        atomicAdd(&t[d], x[s] * dis[s] * dis[d]);
    }
}

// h1[i,j] = relu(t[i]*W1[j] + b1[j])   (layer-1 outer product, fp32)
__global__ void k_h1(const float* __restrict__ t, const float* __restrict__ W1,
                     const float* __restrict__ b1, float* __restrict__ h1, int n, int d) {
    int idx = blockIdx.x * blockDim.x + threadIdx.x;
    if (idx < n * d) {
        int i = idx / d, j = idx - i * d;
        h1[idx] = fmaxf(fmaf(t[i], W1[j], b1[j]), 0.0f);
    }
}

// agg[i,:] = h[i,:] * dis[i]^2   (self-loop term; full overwrite)
__global__ void k_selfloop(const float* __restrict__ h, const float* __restrict__ dis,
                           float* __restrict__ agg, int n, int d) {
    int idx = blockIdx.x * blockDim.x + threadIdx.x;
    if (idx < n * d) {
        int i = idx / d;
        float dd = dis[i];
        agg[idx] = h[idx] * dd * dd;
    }
}

// agg[dst,:] += h[src,:] * norm   (one thread per edge x 4 columns)
__global__ void k_scatter(const float* __restrict__ h, const long long* __restrict__ ei,
                          const float* __restrict__ dis, float* __restrict__ agg,
                          int E, int dq, int d) {
    int idx = blockIdx.x * blockDim.x + threadIdx.x;
    if (idx >= E * dq) return;
    int e = idx / dq;
    int c = (idx - e * dq) * 4;
    int s = (int)ei[e], dst = (int)ei[E + e];
    float nm = dis[s] * dis[dst];
    const float4 v = *(const float4*)(h + (long long)s * d + c);
    float* p = agg + (long long)dst * d + c;
    atomicAdd(p + 0, v.x * nm);
    atomicAdd(p + 1, v.y * nm);
    atomicAdd(p + 2, v.z * nm);
    atomicAdd(p + 3, v.w * nm);
}

__global__ void k_cvt_bf16(const float* __restrict__ a, __bf16* __restrict__ b, int n) {
    int i = blockIdx.x * blockDim.x + threadIdx.x;
    if (i < n) b[i] = (__bf16)a[i];
}

// Wt[n*K + k] = bf16(W[k*Nout + n])  (transpose + convert, per launch)
__global__ void k_wt(const float* __restrict__ W, __bf16* __restrict__ Wt, int K, int Nout) {
    int idx = blockIdx.x * blockDim.x + threadIdx.x;
    if (idx < K * Nout) {
        int k = idx / Nout, n = idx - k * Nout;
        Wt[(long long)n * K + k] = (__bf16)W[idx];
    }
}

__global__ void k_cnt(const long long* __restrict__ batch, float* __restrict__ cnt, int n) {
    int i = blockIdx.x * blockDim.x + threadIdx.x;
    if (i < n) atomicAdd(&cnt[(int)batch[i]], 1.0f);
}

__global__ void k_pool(const float* __restrict__ h, const long long* __restrict__ batch,
                       float* __restrict__ psum, int n, int dq, int d) {
    int idx = blockIdx.x * blockDim.x + threadIdx.x;
    if (idx >= n * dq) return;
    int i = idx / dq;
    int c = (idx - i * dq) * 4;
    int g = (int)batch[i];
    const float4 v = *(const float4*)(h + (long long)i * d + c);
    float* p = psum + g * d + c;
    atomicAdd(p + 0, v.x);
    atomicAdd(p + 1, v.y);
    atomicAdd(p + 2, v.z);
    atomicAdd(p + 3, v.w);
}

__global__ void k_pool_fin(const float* __restrict__ psum, const float* __restrict__ cnt,
                           __bf16* __restrict__ pb, int G, int d) {
    int idx = blockIdx.x * blockDim.x + threadIdx.x;
    if (idx < G * d) {
        int g = idx / d;
        pb[idx] = (__bf16)(psum[idx] / fmaxf(cnt[g], 1.0f));
    }
}

// ---------------- WMMA bf16 GEMM: C = relu?(A @ Wt^T + bias) ----------------
// A:  Mrows x K  row-major bf16
// Wt: Nout  x K  row-major bf16 (i.e. B = Wt^T, K x Nout)
// C:  Mrows x Nout fp32
// block = 256 threads (8 waves); wave tile = 32(M) x 64(N); block tile = 256 x 64.
// - All waves always valid: out-of-range rows clamp to Mrows-1 for loads
//   (duplicate compute); only epilogue stores are range-guarded -> the k-loop
//   is branch-free (no exec toggling around WMMA).
// - B tiles are staged global->LDS with ASYNC loads into ping-pong buffers:
//   tile k+1 is in flight while tile k feeds the WMMAs (in-order async
//   completion => s_wait_asynccnt 1 retires the oldest outstanding load).
__global__ __launch_bounds__(256) void k_gemm_bf16(
    const __bf16* __restrict__ A, const __bf16* __restrict__ Wt,
    const float* __restrict__ bias, float* __restrict__ C,
    int Mrows, int K, int Nout, int relu)
{
    __shared__ __attribute__((aligned(16))) __bf16 lb[2 * 64 * 32];  // 2 x (64 cols x 32 k)
    const int tid  = threadIdx.x;
    const int lane = tid & 31;
    const int w    = tid >> 5;
    const int n0   = blockIdx.x * 64;
    const int mb   = blockIdx.y * 256 + w * 32;
    const int hi = lane >> 4;      // 0: K 0..7/16..23, 1: K 8..15/24..31
    const int nl = lane & 15;

    int m0 = mb + nl;       if (m0 >= Mrows) m0 = Mrows - 1;
    int m1 = mb + 16 + nl;  if (m1 >= Mrows) m1 = Mrows - 1;
    const __bf16* pA0 = A + (long long)m0 * K + hi * 8;
    const __bf16* pA1 = A + (long long)m1 * K + hi * 8;

    // cooperative B-tile staging: thread t covers 8 bf16 of row (n0 + t/4)
    const int ln = tid >> 2;          // 0..63 local column
    const int lk = (tid & 3) * 8;     // 0,8,16,24
    const __bf16* pW = Wt + (long long)(n0 + ln) * K + lk;

    v8f acc[2][4] = {};

    auto compute_tile = [&](const __bf16* base) {
        Frag16 a0, a1;
        a0.u[0] = ld128(pA0);          // K offsets {0..7} or {8..15}
        a0.u[1] = ld128(pA0 + 16);     // K offsets {16..23} or {24..31}
        a1.u[0] = ld128(pA1);
        a1.u[1] = ld128(pA1 + 16);
        __builtin_prefetch(pA0 + 64, 0, 3);
        __builtin_prefetch(pA1 + 64, 0, 3);
        pA0 += 32;
        pA1 += 32;
#pragma unroll
        for (int s = 0; s < 4; ++s) {
            const __bf16* pB = base + (s * 16 + nl) * 32 + hi * 8;
            Frag16 b;
            b.u[0] = ld128(pB);
            b.u[1] = ld128(pB + 16);
            acc[0][s] = __builtin_amdgcn_wmma_f32_16x16x32_bf16(
                false, a0.v, false, b.v, (short)0, acc[0][s], false, false);
            acc[1][s] = __builtin_amdgcn_wmma_f32_16x16x32_bf16(
                false, a1.v, false, b.v, (short)0, acc[1][s], false, false);
        }
    };

#if USE_ASYNC_LDS
    // low 32 bits of the generic shared address are the LDS byte offset
    const unsigned off0 = (unsigned)(size_t)(&lb[ln * 32 + lk]);
    const unsigned off1 = off0 + 64 * 32 * 2;
    const __bf16* buf0 = &lb[0];
    const __bf16* buf1 = &lb[64 * 32];
    const int ntiles = K >> 5;                 // K in {256,512,1024} -> even

    async_b128(off0, pW); pW += 32;            // tile 0 -> buf0
    for (int ti = 0; ti < ntiles; ti += 2) {
        async_b128(off1, pW); pW += 32;        // tile ti+1 -> buf1
        asm volatile("s_wait_asynccnt 1" ::: "memory");   // tile ti landed
        __syncthreads();
        compute_tile(buf0);
        __syncthreads();                       // done reading buf0

        if (ti + 2 < ntiles) {
            async_b128(off0, pW); pW += 32;    // tile ti+2 -> buf0
            asm volatile("s_wait_asynccnt 1" ::: "memory");  // tile ti+1 landed
        } else {
            asm volatile("s_wait_asynccnt 0" ::: "memory");
        }
        __syncthreads();
        compute_tile(buf1);
        __syncthreads();                       // done reading buf1
    }
#else
    for (int k0 = 0; k0 < K; k0 += 32) {
        *(v4u*)(&lb[ln * 32 + lk]) = ld128(pW);
        pW += 32;
        __syncthreads();
        compute_tile(&lb[0]);
        __syncthreads();
    }
#endif

    // epilogue: bias + optional relu, range-guarded stores
#pragma unroll
    for (int t = 0; t < 2; ++t) {
        const int rbase = mb + t * 16 + hi * 8;   // VGPR r -> row rbase + r
#pragma unroll
        for (int s = 0; s < 4; ++s) {
            int col = n0 + s * 16 + nl;
            float bv = bias[col];
#pragma unroll
            for (int r = 0; r < 8; ++r) {
                int row = rbase + r;
                if (row < Mrows) {
                    float v = acc[t][s][r] + bv;
                    if (relu) v = fmaxf(v, 0.0f);
                    C[(long long)row * Nout + col] = v;
                }
            }
        }
    }
}

// ---------------- host-side orchestration ----------------

static inline int cdiv(int a, int b) { return (a + b - 1) / b; }

extern "C" void kernel_launch(void* const* d_in, const int* in_sizes, int n_in,
                              void* d_out, int out_size, void* d_ws, size_t ws_size,
                              hipStream_t stream) {
    const float*     x     = (const float*)d_in[0];
    const long long* ei    = (const long long*)d_in[1];
    const long long* batch = (const long long*)d_in[2];
    const float* W1 = (const float*)d_in[3];
    const float* b1 = (const float*)d_in[4];
    const float* W2 = (const float*)d_in[5];
    const float* b2 = (const float*)d_in[6];
    const float* W3 = (const float*)d_in[7];
    const float* b3 = (const float*)d_in[8];
    const float* Wo = (const float*)d_in[9];
    const float* bo = (const float*)d_in[10];
    float* out = (float*)d_out;

    const int N  = in_sizes[0];        // x is (N,1)
    const int E  = in_sizes[1] / 2;    // edge_index is (2,E)
    const int G  = 16;
    const int D1 = 256, D2 = 512, D3 = 1024, DOUT = 1024;

    // ---- workspace layout (256B-aligned bump allocator, with aliasing) ----
    char* p = (char*)d_ws;
    auto alloc = [&](size_t bytes) -> void* {
        void* r = (void*)p;
        p += (bytes + 255) & ~(size_t)255;
        return r;
    };
    float*  deg  = (float*)alloc((size_t)N * 4);
    float*  dis  = (float*)alloc((size_t)N * 4);
    float*  t    = (float*)alloc((size_t)N * 4);
    float*  cnt  = (float*)alloc((size_t)G * 4);
    float*  psum = (float*)alloc((size_t)G * D3 * 4);
    __bf16* pb   = (__bf16*)alloc((size_t)G * D3 * 2);
    __bf16* W2t  = (__bf16*)alloc((size_t)D1 * D2 * 2);
    __bf16* W3t  = (__bf16*)alloc((size_t)D2 * D3 * 2);
    __bf16* Wot  = (__bf16*)alloc((size_t)D3 * DOUT * 2);
    float*  h1f  = (float*)alloc((size_t)N * D1 * 4);   // aliased by a2b later
    float*  agg2 = (float*)alloc((size_t)N * D1 * 4);   // aliased by a3b later
    float*  h2f  = (float*)alloc((size_t)N * D2 * 4);   // h3 spans h2f..agg3
    float*  agg3 = (float*)alloc((size_t)N * D2 * 4);
    __bf16* a2b = (__bf16*)h1f;    // h1f dead after agg2 is built
    __bf16* a3b = (__bf16*)agg2;   // agg2 dead after a2b conversion
    float*  h3  = h2f;             // h2f+agg3 contiguous = exactly N*D3 floats
    (void)ws_size; (void)n_in; (void)out_size; (void)agg3;

    const int T = 256;

    // ---- degree / normalization ----
    k_fill_f32<<<cdiv(N, T), T, 0, stream>>>(deg, 1.0f, N);          // self-loops
    k_deg_edges<<<cdiv(E, T), T, 0, stream>>>(ei, deg, E);
    k_rsqrt<<<cdiv(N, T), T, 0, stream>>>(deg, dis, N);

    // ---- t = A_hat @ x  (scalar propagate, layer-1 input dim = 1) ----
    k_t_self<<<cdiv(N, T), T, 0, stream>>>(x, dis, t, N);
    k_t_edges<<<cdiv(E, T), T, 0, stream>>>(x, ei, dis, t, E);

    // ---- weight transpose+convert to bf16 (per-launch, stateless) ----
    k_wt<<<cdiv(D1 * D2, T), T, 0, stream>>>(W2, W2t, D1, D2);
    k_wt<<<cdiv(D2 * D3, T), T, 0, stream>>>(W3, W3t, D2, D3);
    k_wt<<<cdiv(D3 * DOUT, T), T, 0, stream>>>(Wo, Wot, D3, DOUT);

    // ---- layer 1: h1 = relu(t * W1 + b1) ----
    k_h1<<<cdiv(N * D1, T), T, 0, stream>>>(t, W1, b1, h1f, N, D1);

    // ---- layer 2: a2 = A_hat @ h1 ; h2 = relu(a2 @ W2 + b2) ----
    k_selfloop<<<cdiv(N * D1, T), T, 0, stream>>>(h1f, dis, agg2, N, D1);
    k_scatter<<<cdiv(E * (D1 / 4), T), T, 0, stream>>>(h1f, ei, dis, agg2, E, D1 / 4, D1);
    k_cvt_bf16<<<cdiv(N * D1, T), T, 0, stream>>>(agg2, a2b, N * D1);
    {
        dim3 g(D2 / 64, cdiv(N, 256));
        k_gemm_bf16<<<g, 256, 0, stream>>>(a2b, W2t, b2, h2f, N, D1, D2, 1);
    }

    // ---- layer 3: a3 = A_hat @ h2 ; h3 = relu(a3 @ W3 + b3) ----
    k_selfloop<<<cdiv(N * D2, T), T, 0, stream>>>(h2f, dis, agg3, N, D2);
    k_scatter<<<cdiv(E * (D2 / 4), T), T, 0, stream>>>(h2f, ei, dis, agg3, E, D2 / 4, D2);
    k_cvt_bf16<<<cdiv(N * D2, T), T, 0, stream>>>(agg3, a3b, N * D2);
    {
        dim3 g(D3 / 64, cdiv(N, 256));
        k_gemm_bf16<<<g, 256, 0, stream>>>(a3b, W3t, b3, h3, N, D2, D3, 1);
    }

    // ---- global mean pool ----
    k_fill_f32<<<cdiv(G * D3, T), T, 0, stream>>>(psum, 0.0f, G * D3);
    k_fill_f32<<<1, T, 0, stream>>>(cnt, 0.0f, G);
    k_cnt<<<cdiv(N, T), T, 0, stream>>>(batch, cnt, N);
    k_pool<<<cdiv(N * (D3 / 4), T), T, 0, stream>>>(h3, batch, psum, N, D3 / 4, D3);
    k_pool_fin<<<cdiv(G * D3, T), T, 0, stream>>>(psum, cnt, pb, G, D3);

    // ---- output projection: out = pooled @ Wo + bo  (16 rows = one M tile) ----
    {
        dim3 g(DOUT / 64, cdiv(G, 256));
        k_gemm_bf16<<<g, 256, 0, stream>>>(pb, Wot, bo, out, G, D3, DOUT, 0);
    }
}
